// model_encdec_75101798138156
// MI455X (gfx1250) — matrix operand order repaired and checked by reading, not compiled, not run on hardware
//
#include <hip/hip_runtime.h>

typedef __attribute__((ext_vector_type(16))) _Float16 v16h;
typedef __attribute__((ext_vector_type(8)))  _Float16 v8h;
typedef __attribute__((ext_vector_type(8)))  float    v8f;

#define NW 4            // waves per workgroup (wave32), 16 batch rows per wave
#define NFRAG_LDS 48    // encWih(9) + encWhh(18) + decWhh(18) + fc(2) + conv(1)
#define NFRAG_ALL 66    // + decWih(18) kept in global/L2
#define FRAG_F16  512   // 32 lanes x 16 f16 per B fragment

union U16 { v16h v; v8h h[2]; };

__device__ __forceinline__ v8f zf8(){ v8f z;
#pragma unroll
  for (int i=0;i<8;++i) z[i]=0.f; return z; }

__device__ __forceinline__ v8h zh8(){ v8h z;
#pragma unroll
  for (int i=0;i<8;++i) z[i]=(_Float16)0.f; return z; }

__device__ __forceinline__ v8f wmma16(v16h a, v16h b, v8f c){
  // D = A(16x32 f16) x B(32x16 f16) + C(16x16 f32)
  return __builtin_amdgcn_wmma_f32_16x16x32_f16(false, a, false, b, (short)0, c, false, false);
}

// rcp-based activations: saturate correctly at +-inf (rcp(inf)=0), no clamps needed
__device__ __forceinline__ float sigm(float x){
  return __builtin_amdgcn_rcpf(1.f + __expf(-x));
}
__device__ __forceinline__ float tanh_f(float x){
  return __builtin_fmaf(2.f, __builtin_amdgcn_rcpf(1.f + __expf(-2.f*x)), -1.f);
}

// Launder the per-lane byte offset ONCE per loop iteration: all weight-fragment
// ds_loads share the laundered base register (fragment index folds into the
// 16-bit ds_load immediate), so loads can't be hoisted/CSE'd across iterations
// (fresh SSA value each iteration -> stays under 256 VGPRs) at a cost of zero
// real instructions per iteration.
__device__ __forceinline__ const _Float16* launder_base(const _Float16* base, int lane){
  int boff = lane * 32;                      // bytes: 16 f16 per lane
  asm volatile("" : "+v"(boff));
  return (const _Float16*)((const char*)base + boff);
}
// B fragment at laundered base + fragment index (offset becomes an immediate).
__device__ __forceinline__ v16h bfrag_at(const _Float16* lbase, int fragIdx){
  U16 u;
  const v8h* p = (const v8h*)(lbase + fragIdx*FRAG_F16);
  u.h[0] = p[0]; u.h[1] = p[1];
  return u.v;
}
// B fragment, plain: compiler may keep it register-resident (single hot fragments).
__device__ __forceinline__ v16h bfrag(const _Float16* base, int lane){
  U16 u;
  const v8h* p = (const v8h*)(base + lane*16);
  u.h[0] = p[0]; u.h[1] = p[1];
  return u.v;
}
// A fragment from 16x64 f16 row-major LDS tile, K-chunk c (k = 32c..32c+31)
// lanes 0-15: row=lane, k {0..7,16..23}; lanes 16-31: row=lane-16, k {8..15,24..31}
__device__ __forceinline__ v16h afrag_h(const _Float16* hst, int lane, int c){
  U16 u;
  const int r  = lane & 15;
  const int kb = c*32 + ((lane >> 4) << 3);
  u.h[0] = *(const v8h*)(hst + r*64 + kb);
  u.h[1] = *(const v8h*)(hst + r*64 + kb + 16);
  return u.v;
}
// A fragment from 16x16 f16 tile, K padded to 32 with zeros
__device__ __forceinline__ v16h afrag_emb(const _Float16* emb, int lane){
  U16 u;
  const int r  = lane & 15;
  const int kb = (lane >> 4) << 3;
  u.h[0] = *(const v8h*)(emb + r*16 + kb);
  u.h[1] = zh8();
  return u.v;
}

// GRU gate math on D-layout accumulators; writes new h (f16) back to the LDS tile.
// rz[6]: combined gx+gh for r (0..2) and z (3..5) tiles. xn/hn: n-gate pieces (nullable).
__device__ __forceinline__ void gru_gates(
    v8f (&h)[3], const v8f (&rz)[6], const v8f* xn, const v8f* hn_,
    _Float16* hst, const float (&brz)[6], const float (&bin)[3], const float (&bhn)[3],
    int col, int mb)
{
#pragma unroll
  for (int i=0;i<3;++i){
#pragma unroll
    for (int e=0;e<8;++e){
      const float r  = sigm(rz[i][e]   + brz[i]);
      const float z  = sigm(rz[i+3][e] + brz[i+3]);
      const float gn = (xn ? xn[i][e] : 0.f) + bin[i] + r*((hn_ ? hn_[i][e] : 0.f) + bhn[i]);
      const float n  = tanh_f(gn);
      const float hv = __builtin_fmaf(z, h[i][e] - n, n);   // (1-z)*n + z*h
      h[i][e] = hv;
      hst[(e+mb)*64 + i*16 + col] = (_Float16)hv;
    }
  }
}

// ============ prep: pack weights into WMMA B-fragment layout (f16) + f32 tables ============
__global__ void pack_weights(
    const float* __restrict__ conv_w, const float* __restrict__ conv_b,
    const float* __restrict__ eWih,  const float* __restrict__ eWhh,
    const float* __restrict__ ebih,  const float* __restrict__ ebhh,
    const float* __restrict__ dWih,  const float* __restrict__ dWhh,
    const float* __restrict__ dbih,  const float* __restrict__ dbhh,
    const float* __restrict__ fcW,   const float* __restrict__ fcb,
    _Float16* __restrict__ pack16, float* __restrict__ pack32)
{
  const int idx = blockIdx.x*blockDim.x + threadIdx.x;
  const int NF16 = NFRAG_ALL * FRAG_F16;
  if (idx < NF16){
    const int f = idx >> 9, r = idx & 511;
    const int lane = r >> 4, e = r & 15;
    const int k = ((lane >> 4) << 3) + (e < 8 ? e : e + 8);   // 0..31 within K-chunk
    const int n = lane & 15;
    float v = 0.f;
    if (f < 9){                       // enc_Wih [144][16], K=16 padded to 32
      const int ng = f*16 + n;
      if (k < 16) v = eWih[ng*16 + k];
    } else if (f < 27){               // enc_Whh [144][48], K padded to 64
      const int ix=f-9, tile=ix>>1, c=ix&1, ng=tile*16+n, kg=c*32+k;
      if (kg < 48) v = eWhh[ng*48 + kg];
    } else if (f < 45){               // dec_Whh
      const int ix=f-27, tile=ix>>1, c=ix&1, ng=tile*16+n, kg=c*32+k;
      if (kg < 48) v = dWhh[ng*48 + kg];
    } else if (f < 47){               // fc_W [3][48] in cols 0..2 of one N-tile
      const int c=f-45, kg=c*32+k;
      if (n < 3 && kg < 48) v = fcW[n*48 + kg];
    } else if (f == 47){              // conv_w as B: row j=kk*3+c (9 rows), col = out-ch
      if (k < 9) v = conv_w[n*9 + (k%3)*3 + (k/3)];
    } else {                          // dec_Wih
      const int ix=f-48, tile=ix>>1, c=ix&1, ng=tile*16+n, kg=c*32+k;
      if (kg < 48) v = dWih[ng*48 + kg];
    }
    pack16[idx] = (_Float16)v;
  } else if (idx < NF16 + 740){
    const int j = idx - NF16;
    float v;
    if      (j < 144) v = conv_w[j];
    else if (j < 160) v = conv_b[j-144];
    else if (j < 304) v = ebih[j-160];
    else if (j < 448) v = ebhh[j-304];
    else if (j < 592) v = dbih[j-448];
    else if (j < 736) v = dbhh[j-592];
    else              v = (j-736 < 3) ? fcb[j-736] : 0.f;
    pack32[j] = v;
  }
}

// ============ fused conv + encoder GRU + decoder GRU + fc ============
__global__ __launch_bounds__(NW*32)
void fused_encdec(const float* __restrict__ past,
                  const _Float16* __restrict__ pack16,
                  const float* __restrict__ pack32,
                  float* __restrict__ out, int B)
{
  __shared__ __align__(32) _Float16 sFrag[NFRAG_LDS * FRAG_F16]; // 49152 B
  __shared__            float     sF32[740];                     //  2960 B
  __shared__ __align__(32) _Float16 sH[NW][16*64];               //  8192 B
  __shared__ __align__(32) _Float16 sEmb[NW][16*16];             //  2048 B

  { // cooperative weight staging
    const unsigned* s = (const unsigned*)pack16;
    unsigned* d = (unsigned*)sFrag;
    for (int i=threadIdx.x; i < NFRAG_LDS*FRAG_F16/2; i += blockDim.x) d[i] = s[i];
    const unsigned* s2 = (const unsigned*)pack32;
    unsigned* d2 = (unsigned*)sF32;
    for (int i=threadIdx.x; i < 740; i += blockDim.x) d2[i] = s2[i];
  }
  __syncthreads();

  const int lane = threadIdx.x & 31;
  const int wv   = threadIdx.x >> 5;
  const int b0   = (blockIdx.x*NW + wv)*16;
  if (b0 >= B) return;

  const int col = lane & 15;
  const int mb  = (lane >> 4) << 3;

  _Float16* hst = sH[wv];
  _Float16* emb = sEmb[wv];
  const _Float16* decWih = pack16 + NFRAG_LDS*FRAG_F16;  // streamed from global/L2

  for (int i=lane; i<16*64; i+=32) hst[i] = (_Float16)0.f;  // h0 = 0 (incl. K-pad)

  // per-lane scalars: conv bias (per out-channel) and encoder gate biases
  const float cbias = sF32[144 + col];
  float ebrz[6], ebin[3], ebhn[3];
#pragma unroll
  for (int i=0;i<3;++i){
    ebrz[i]   = sF32[160 +      i*16+col] + sF32[304 +      i*16+col];
    ebrz[i+3] = sF32[160 + 48 + i*16+col] + sF32[304 + 48 + i*16+col];
    ebin[i]   = sF32[160 + 96 + i*16+col];
    ebhn[i]   = sF32[304 + 96 + i*16+col];
  }

  // conv sliding window: every lane tracks row = lane&15 (hi half supplies k=8)
  const float* pp = past + (size_t)(b0 + col)*60;   // [20][3]
  float win[9];
#pragma unroll
  for (int j=0;j<3;++j) win[j] = 0.f;               // t=-1 pad
#pragma unroll
  for (int j=0;j<6;++j) win[3+j] = pp[j];           // t=0, t=1

  v8f h[3];
#pragma unroll
  for (int i=0;i<3;++i) h[i] = zf8();

  // single conv B-fragment stays register-resident (8 VGPRs, reused 20x)
  const v16h convB = bfrag(sFrag + 47*FRAG_F16, lane);

  // ================= encoder =================
#pragma unroll 1
  for (int t=0;t<20;++t){
    // fresh laundered weight base for this iteration (blocks cross-iteration CSE)
    const _Float16* wb = launder_base(sFrag, lane);

    // conv1d(3->16,k=3,pad=1) as one WMMA: A = window rows (K=9 pad 32)
    U16 ua;
    {
      const bool hi = lane >= 16;
      v8h lo;
      lo[0] = (_Float16)(hi ? win[8] : win[0]);
#pragma unroll
      for (int j=1;j<8;++j) lo[j] = (_Float16)(hi ? 0.f : win[j]);
      ua.h[0] = lo; ua.h[1] = zh8();
    }
    v8f ec = wmma16(ua.v, convB, zf8());
    // bias + relu -> emb tile (D-layout scatter to row-major f16)
#pragma unroll
    for (int e=0;e<8;++e)
      emb[(e+mb)*16 + col] = (_Float16)fmaxf(ec[e] + cbias, 0.f);
    // slide window
#pragma unroll
    for (int j=0;j<6;++j) win[j] = win[j+3];
#pragma unroll
    for (int c=0;c<3;++c) win[6+c] = (t+2<20) ? pp[(t+2)*3+c] : 0.f;

    // r/z: gx+gh fused into one accumulator set; n: gx and gh kept separate
    v8f rz[6], xn[3], hn[3];
    {
      v16h ax = afrag_emb(emb, lane);
#pragma unroll
      for (int tt=0;tt<6;++tt) rz[tt] = wmma16(ax, bfrag_at(wb, tt), zf8());
#pragma unroll
      for (int i=0;i<3;++i)    xn[i]  = wmma16(ax, bfrag_at(wb, 6+i), zf8());
    }
#pragma unroll
    for (int i=0;i<3;++i) hn[i] = zf8();
#pragma unroll
    for (int c=0;c<2;++c){
      v16h ah = afrag_h(hst, lane, c);
#pragma unroll
      for (int tt=0;tt<6;++tt)
        rz[tt] = wmma16(ah, bfrag_at(wb, 9 + tt*2 + c), rz[tt]);
#pragma unroll
      for (int i=0;i<3;++i)
        hn[i] = wmma16(ah, bfrag_at(wb, 9 + (6+i)*2 + c), hn[i]);
    }
    gru_gates(h, rz, xn, hn, hst, ebrz, ebin, ebhn, col, mb);
  }

  // write state_past output (f32, offset B*60)
  {
    const size_t so = (size_t)B*60;
#pragma unroll
    for (int i=0;i<3;++i)
#pragma unroll
      for (int e=0;e<8;++e)
        out[so + (size_t)(b0 + e + mb)*48 + i*16 + col] = h[i][e];
  }

  // decoder gate biases
  float dbrz[6], dbin[3], dbhn[3];
#pragma unroll
  for (int i=0;i<3;++i){
    dbrz[i]   = sF32[448 +      i*16+col] + sF32[592 +      i*16+col];
    dbrz[i+3] = sF32[448 + 48 + i*16+col] + sF32[592 + 48 + i*16+col];
    dbin[i]   = sF32[448 + 96 + i*16+col];
    dbhn[i]   = sF32[592 + 96 + i*16+col];
  }
  const float fcb_l = sF32[736 + (col < 3 ? col : 0)];

  // fc B-fragments stay register-resident (16 VGPRs, reused 20x)
  const v16h fcB0 = bfrag(sFrag + 45*FRAG_F16, lane);
  const v16h fcB1 = bfrag(sFrag + 46*FRAG_F16, lane);

  auto do_fc = [&](int t){
    v8f fa = wmma16(afrag_h(hst, lane, 0), fcB0, zf8());
    fa     = wmma16(afrag_h(hst, lane, 1), fcB1, fa);
    if (col < 3){
#pragma unroll
      for (int e=0;e<8;++e)
        out[((size_t)(b0 + e + mb)*20 + t)*3 + col] = fa[e] + fcb_l;
    }
  };

  // ---- decoder step 0 (peeled): x = state_past (still in hst), h0 = 0 -> gh = 0
  {
    v8f rz[6], xn[3];
#pragma unroll
    for (int i=0;i<6;++i) rz[i] = zf8();
#pragma unroll
    for (int i=0;i<3;++i) xn[i] = zf8();
#pragma unroll
    for (int c=0;c<2;++c){
      v16h as = afrag_h(hst, lane, c);
#pragma unroll
      for (int tt=0;tt<6;++tt)
        rz[tt] = wmma16(as, bfrag(decWih + (tt*2 + c)*FRAG_F16, lane), rz[tt]);
#pragma unroll
      for (int i=0;i<3;++i)
        xn[i] = wmma16(as, bfrag(decWih + ((6+i)*2 + c)*FRAG_F16, lane), xn[i]);
    }
#pragma unroll
    for (int i=0;i<3;++i) h[i] = zf8();     // h0 = 0
    gru_gates(h, rz, xn, (const v8f*)nullptr, hst, dbrz, dbin, dbhn, col, mb);
    do_fc(0);
  }

  // ---- decoder steps 1..19: x = 0 -> gx is bias-only
#pragma unroll 1
  for (int t=1;t<20;++t){
    const _Float16* wb = launder_base(sFrag + 27*FRAG_F16, lane);
    v8f rz[6], hn[3];
#pragma unroll
    for (int i=0;i<6;++i) rz[i] = zf8();
#pragma unroll
    for (int i=0;i<3;++i) hn[i] = zf8();
#pragma unroll
    for (int c=0;c<2;++c){
      v16h ah = afrag_h(hst, lane, c);
#pragma unroll
      for (int tt=0;tt<6;++tt)
        rz[tt] = wmma16(ah, bfrag_at(wb, tt*2 + c), rz[tt]);
#pragma unroll
      for (int i=0;i<3;++i)
        hn[i] = wmma16(ah, bfrag_at(wb, (6+i)*2 + c), hn[i]);
    }
    gru_gates(h, rz, (const v8f*)nullptr, hn, hst, dbrz, dbin, dbhn, col, mb);
    do_fc(t);
  }
}

extern "C" void kernel_launch(void* const* d_in, const int* in_sizes, int n_in,
                              void* d_out, int out_size, void* d_ws, size_t ws_size,
                              hipStream_t stream)
{
  const float* past   = (const float*)d_in[0];
  const float* conv_w = (const float*)d_in[1];
  const float* conv_b = (const float*)d_in[2];
  const float* eWih   = (const float*)d_in[3];
  const float* eWhh   = (const float*)d_in[4];
  const float* ebih   = (const float*)d_in[5];
  const float* ebhh   = (const float*)d_in[6];
  const float* dWih   = (const float*)d_in[7];
  const float* dWhh   = (const float*)d_in[8];
  const float* dbih   = (const float*)d_in[9];
  const float* dbhh   = (const float*)d_in[10];
  const float* fcW    = (const float*)d_in[11];
  const float* fcb    = (const float*)d_in[12];

  _Float16* pack16 = (_Float16*)d_ws;
  float*    pack32 = (float*)((char*)d_ws + NFRAG_ALL*FRAG_F16*2);
  float*    out    = (float*)d_out;
  const int B = in_sizes[0] / 60;    // past = [B,20,3]

  const int prep_elems = NFRAG_ALL*FRAG_F16 + 740;
  pack_weights<<<(prep_elems + 255)/256, 256, 0, stream>>>(
      conv_w, conv_b, eWih, eWhh, ebih, ebhh, dWih, dWhh, dbih, dbhh, fcW, fcb,
      pack16, pack32);

  const int rowsPerBlock = NW*16;    // 64 batch rows per workgroup
  fused_encdec<<<(B + rowsPerBlock - 1)/rowsPerBlock, NW*32, 0, stream>>>(
      past, pack16, pack32, out, B);
}
